// SaccadicGPT2Block_19499151523868
// MI455X (gfx1250) — matrix operand order repaired and checked by reading, not compiled, not run on hardware
//
#include <hip/hip_runtime.h>
#include <hip/hip_bf16.h>

// ---------------------------------------------------------------------------
// Types
// ---------------------------------------------------------------------------
typedef __bf16 bf16_t;
typedef __attribute__((ext_vector_type(16))) __bf16 v16bf;
typedef __attribute__((ext_vector_type(8)))  float  v8f;
typedef __attribute__((ext_vector_type(2)))  unsigned int u32x2;
typedef __attribute__((ext_vector_type(4)))  unsigned int u32x4;
typedef __attribute__((ext_vector_type(4)))  int          i32x4;
typedef __attribute__((ext_vector_type(8)))  int          i32x8;

union Frag16 {                 // 32 bytes: one 16-bit A or B WMMA fragment
    v16bf  v;
    __bf16 e[16];
    u32x2  u2[4];
    u32x4  u4[2];
};

__device__ __forceinline__ v8f wmma_bf16(v16bf a, v16bf b, v8f c) {
    // D = A(16x32 bf16) x B(32x16 bf16) + C(16x16 f32)
    return __builtin_amdgcn_wmma_f32_16x16x32_bf16(
        /*neg_a=*/false, a, /*neg_b=*/false, b,
        /*c_mod=*/(short)0, c, /*reuse_a=*/false, /*reuse_b=*/false);
}

// Problem constants
#define DMODEL 1024
#define SEQL   2048
#define BATCH  4
#define NHEAD  16
#define HDIM   64
#define NWIN   5
#define WLEN   128
#define MKV    640          // NWIN * WLEN
#define NBLK   64           // SEQL / 32
#define DFFN   4096

// ---------------------------------------------------------------------------
// TDM (Tensor Data Mover) support: DMA a 2-D [rows x 32] bf16 tile from a
// row-major matrix into LDS with a padded row stride of 40 elements (80 B),
// matching the WMMA fragment layout below. Guarded so the manual staging path
// remains as a fallback on toolchains without the builtin.
// ---------------------------------------------------------------------------
#if defined(__has_builtin)
#if __has_builtin(__builtin_amdgcn_tensor_load_to_lds) && __has_builtin(__builtin_amdgcn_s_wait_tensorcnt)
#define HAVE_TDM 1
#endif
#endif

__device__ __forceinline__ unsigned lds_byte_off(const void* p) {
    // generic -> LDS address space; AS3 pointers are LDS byte offsets
    return (unsigned)(unsigned long long)(__attribute__((address_space(3))) const char*)p;
}

#ifdef HAVE_TDM
// Issue one TDM load: tile [32 cols x nrows] of bf16 from row-major src
// (row stride = lda elements) into LDS at ldsOff, LDS row stride 80 bytes.
__device__ __forceinline__ void tdm_tile_load(const bf16_t* srcTile, unsigned ldsOff,
                                              int lda, int dimRows, int nrows) {
    unsigned long long ga = (unsigned long long)srcTile;
    u32x4 g0;
    g0[0] = 1u;                                       // count=1, user descriptor
    g0[1] = ldsOff;                                   // lds_addr (bytes)
    g0[2] = (unsigned)(ga & 0xFFFFFFFFull);           // global_addr[31:0]
    g0[3] = (unsigned)((ga >> 32) & 0x01FFFFFFull)    // global_addr[56:32]
          | 0x80000000u;                              // type = 2 ("image")
    // group1: data_size=1 (2B), pad_enable, pad_interval=3 (16 DW = 64 B),
    // pad_amount=3 (4 DW = 16 B) -> LDS row stride 80 B
    i32x8 g1;
    g1[0] = (1 << 16) | (1 << 20) | (3 << 22) | (3 << 25);
    g1[1] = (lda & 0xFFFF) << 16;                     // tensor_dim0[15:0]
    g1[2] = ((lda >> 16) & 0xFFFF)                    // tensor_dim0[31:16]
          | ((dimRows & 0xFFFF) << 16);               // tensor_dim1[15:0]
    g1[3] = ((dimRows >> 16) & 0xFFFF)                // tensor_dim1[31:16]
          | (32 << 16);                               // tile_dim0 = 32 elems
    g1[4] = nrows & 0xFFFF;                           // tile_dim1 rows, tile_dim2=0
    g1[5] = lda;                                      // tensor_dim0_stride[31:0]
    g1[6] = 0;                                        // stride[47:32]=0, dim1_stride lo
    g1[7] = 0;
    i32x4 z4 = {0, 0, 0, 0};
#if __clang_major__ >= 23
    i32x8 z8 = {0, 0, 0, 0, 0, 0, 0, 0};
    __builtin_amdgcn_tensor_load_to_lds(g0, g1, z4, z4, z8, 0);
#else
    __builtin_amdgcn_tensor_load_to_lds(g0, g1, z4, z4, 0);
#endif
}
#endif // HAVE_TDM

// ---------------------------------------------------------------------------
// Weight conversion kernels (fp32 -> bf16, optionally transposed to [N][K])
// ---------------------------------------------------------------------------
__global__ void cvt_bf_kernel(const float* __restrict__ in, bf16_t* __restrict__ out, int n) {
    int i = blockIdx.x * blockDim.x + threadIdx.x;
    int stride = gridDim.x * blockDim.x;
    for (; i < n; i += stride) out[i] = (bf16_t)in[i];
}

// out[n*K + k] = in[k*N + n]   (in is [K][N] row-major)
__global__ void cvt_tr_kernel(const float* __restrict__ in, bf16_t* __restrict__ out, int K, int N) {
    int i = blockIdx.x * blockDim.x + threadIdx.x;
    int stride = gridDim.x * blockDim.x;
    int total = K * N;
    for (; i < total; i += stride) {
        int k = i / N, n = i % N;
        out[(size_t)n * K + k] = (bf16_t)in[i];
    }
}

// ---------------------------------------------------------------------------
// LayerNorm: one block per row of 1024. Writes bf16 normalized row; optionally
// fuses the per-token salience dot-product h . w_sal (fp32).
// ---------------------------------------------------------------------------
__global__ __launch_bounds__(256)
void ln_kernel(const float* __restrict__ x, const float* __restrict__ gam,
               const float* __restrict__ bet, const float* __restrict__ wsal,
               bf16_t* __restrict__ hout, float* __restrict__ sout, int withSal) {
    __shared__ float red[256];
    int row = blockIdx.x;
    int t = threadIdx.x;
    const float* xr = x + (size_t)row * DMODEL;
    float v0 = xr[t], v1 = xr[t + 256], v2 = xr[t + 512], v3 = xr[t + 768];

    red[t] = v0 + v1 + v2 + v3;
    __syncthreads();
    for (int o = 128; o; o >>= 1) { if (t < o) red[t] += red[t + o]; __syncthreads(); }
    float mean = red[0] * (1.0f / DMODEL);
    __syncthreads();

    float d0 = v0 - mean, d1 = v1 - mean, d2 = v2 - mean, d3 = v3 - mean;
    red[t] = d0 * d0 + d1 * d1 + d2 * d2 + d3 * d3;
    __syncthreads();
    for (int o = 128; o; o >>= 1) { if (t < o) red[t] += red[t + o]; __syncthreads(); }
    float rstd = rsqrtf(red[0] * (1.0f / DMODEL) + 1e-5f);
    __syncthreads();

    bf16_t* hr = hout + (size_t)row * DMODEL;
    float h0 = d0 * rstd * gam[t]       + bet[t];
    float h1 = d1 * rstd * gam[t + 256] + bet[t + 256];
    float h2 = d2 * rstd * gam[t + 512] + bet[t + 512];
    float h3 = d3 * rstd * gam[t + 768] + bet[t + 768];
    hr[t]       = (bf16_t)h0;
    hr[t + 256] = (bf16_t)h1;
    hr[t + 512] = (bf16_t)h2;
    hr[t + 768] = (bf16_t)h3;
    if (withSal) {
        float sal = h0 * wsal[t] + h1 * wsal[t + 256] + h2 * wsal[t + 512] + h3 * wsal[t + 768];
        red[t] = sal;
        __syncthreads();
        for (int o = 128; o; o >>= 1) { if (t < o) red[t] += red[t + o]; __syncthreads(); }
        if (t == 0) sout[row] = red[0];
    }
}

// ---------------------------------------------------------------------------
// Saccade selection: block logits (means of salience) + Gumbel perturbation,
// argmax -> window start indices. One tiny block.
// ---------------------------------------------------------------------------
__global__ __launch_bounds__(256)
void select_kernel(const float* __restrict__ sal, const float* __restrict__ gu,
                   const float* __restrict__ bsal, int* __restrict__ startOut) {
    __shared__ float lg[BATCH * NBLK];   // 256
    int t = threadIdx.x;
    {
        int b = t >> 6, nb = t & 63;
        const float* sp = sal + b * SEQL + nb * 32;
        float acc = 0.0f;
        for (int j = 0; j < 32; ++j) acc += sp[j];
        lg[t] = acc * (1.0f / 32.0f) + bsal[0];
    }
    __syncthreads();
    if (t < NWIN * BATCH) {
        int si = t >> 2, b = t & 3;
        const float* gp = gu + (si * BATCH + b) * NBLK;
        float best = -1e30f; int sel = 0;
        for (int nb = 0; nb < NBLK; ++nb) {
            float u = gp[nb];
            float g = -logf(-logf(u + 1e-9f) + 1e-9f);
            float p = lg[b * NBLK + nb] + g;   // TAU == 1
            if (p > best) { best = p; sel = nb; }
        }
        int st = sel * 32 + 16 - WLEN / 2;                // center - W/2
        st = st < 0 ? 0 : (st > (SEQL - WLEN) ? (SEQL - WLEN) : st);
        startOut[t] = st;
    }
}

// ---------------------------------------------------------------------------
// Gather KV windows: kv[b, s*W + w, :] = h[b, start[s,b] + w, :]
// ---------------------------------------------------------------------------
__global__ __launch_bounds__(128)
void gather_kernel(const bf16_t* __restrict__ h, const int* __restrict__ startIn,
                   bf16_t* __restrict__ kv) {
    int row = blockIdx.x;               // b*640 + s*128 + w
    int b = row / MKV, r = row % MKV;
    int si = r >> 7, w = r & 127;
    int st = startIn[si * BATCH + b];
    const u32x4* src = (const u32x4*)(h + ((size_t)(b * SEQL + st + w)) * DMODEL);
    u32x4*       dst = (u32x4*)(kv + (size_t)row * DMODEL);
    dst[threadIdx.x] = src[threadIdx.x];
}

// ---------------------------------------------------------------------------
// Tiled WMMA GEMM: C[M,N] = A[M,K](bf16) x Bw^T + bias, Bw stored [N][K] bf16.
// Block tile 128x128, 8 waves (4 along M x 2 along N), each wave 32x64.
// Tile staging via TDM (tensor_load_to_lds, TENSORcnt) when available.
// epi: 0 = bf16 out, 1 = f32 out + residual, 2 = tanh-GELU bf16 out.
// ---------------------------------------------------------------------------
#define LDT 40   // padded LDS K-stride (keeps 8/16B alignment, avoids conflicts)

__global__ __launch_bounds__(256)
void gemm_kernel(const bf16_t* __restrict__ A, int lda,
                 const bf16_t* __restrict__ Bw,
                 const float* __restrict__ bias,
                 const float* __restrict__ res,
                 float* __restrict__ outF, bf16_t* __restrict__ outB,
                 int M, int N, int K, int ldc, int epi) {
    __shared__ __bf16 At[128 * LDT];
    __shared__ __bf16 Bt[128 * LDT];

    int t = threadIdx.x;
    int nblock = blockIdx.x * 128, mblock = blockIdx.y * 128;

    int lane = t & 31, wave = t >> 5;
    int wm = wave & 3, wn = wave >> 2;
    int arow = wm * 32 + (lane & 15);                   // A frag: lane%16 = M
    int brow = wn * 64 + (lane & 15);                   // B frag: lane%16 = N
    int kbA = (lane >> 4) * 8;                          // A K-subset base
    int kkB = (lane >> 4) * 8;                          // B K-subset base

#ifndef HAVE_TDM
    int lrow = t >> 1, lcol = (t & 1) * 16;             // LDS-fill coords
    const bf16_t* ag = A  + (size_t)(mblock + lrow) * lda + lcol;
    const bf16_t* bg = Bw + (size_t)(nblock + lrow) * K   + lcol;
#else
    unsigned atOff = lds_byte_off(&At[0]);
    unsigned btOff = lds_byte_off(&Bt[0]);
#endif

    v8f Cf[2][4] = {};

    for (int k0 = 0; k0 < K; k0 += 32) {
#ifdef HAVE_TDM
        if (wave == 0) {
            // DMA both tiles: A rows [mblock,+128) cols [k0,+32),
            //                 B rows [nblock,+128) cols [k0,+32)
            tdm_tile_load(A  + (size_t)mblock * lda + k0, atOff, lda, M, 128);
            tdm_tile_load(Bw + (size_t)nblock * K   + k0, btOff, K,   N, 128);
            __builtin_amdgcn_s_wait_tensorcnt(0);
        }
        __syncthreads();
#else
        {   // global -> LDS (each thread: 16 bf16 of A and of B)
            const u32x4* as = (const u32x4*)(ag + k0);
            u32x4* ad = (u32x4*)(&At[lrow * LDT + lcol]);
            ad[0] = as[0]; ad[1] = as[1];
            const u32x4* bs = (const u32x4*)(bg + k0);
            u32x4* bd = (u32x4*)(&Bt[lrow * LDT + lcol]);
            bd[0] = bs[0]; bd[1] = bs[1];
        }
        __syncthreads();
#endif

        Frag16 fa0, fa1;
        {   // A frags (K-interleaved per ISA layout: {kb..kb+7, kb+16..kb+23})
            const __bf16* ar0 = &At[arow * LDT];
            const __bf16* ar1 = &At[(arow + 16) * LDT];
            fa0.u2[0] = *(const u32x2*)(ar0 + kbA);
            fa0.u2[1] = *(const u32x2*)(ar0 + kbA + 4);
            fa0.u2[2] = *(const u32x2*)(ar0 + kbA + 16);
            fa0.u2[3] = *(const u32x2*)(ar0 + kbA + 20);
            fa1.u2[0] = *(const u32x2*)(ar1 + kbA);
            fa1.u2[1] = *(const u32x2*)(ar1 + kbA + 4);
            fa1.u2[2] = *(const u32x2*)(ar1 + kbA + 16);
            fa1.u2[3] = *(const u32x2*)(ar1 + kbA + 20);
        }
#pragma unroll
        for (int fn = 0; fn < 4; ++fn) {
            Frag16 fb;
            const __bf16* br = &Bt[(brow + fn * 16) * LDT + kkB];
            fb.u4[0] = *(const u32x4*)br;
            fb.u4[1] = *(const u32x4*)(br + 16);
            Cf[0][fn] = wmma_bf16(fa0.v, fb.v, Cf[0][fn]);
            Cf[1][fn] = wmma_bf16(fa1.v, fb.v, Cf[1][fn]);
        }
        __syncthreads();
    }

    // epilogue: C/D layout -> lane%16 = N, rows (lane/16)*8 + r
    int colbase = nblock + wn * 64;
    int rowbase = mblock + wm * 32 + (lane >> 4) * 8;
#pragma unroll
    for (int fm = 0; fm < 2; ++fm) {
#pragma unroll
        for (int fn = 0; fn < 4; ++fn) {
            int col = colbase + fn * 16 + (lane & 15);
            float bv = bias[col];
            int r0 = rowbase + fm * 16;
#pragma unroll
            for (int r = 0; r < 8; ++r) {
                size_t idx = (size_t)(r0 + r) * ldc + col;
                float val = Cf[fm][fn][r] + bv;
                if (epi == 0) {
                    outB[idx] = (bf16_t)val;
                } else if (epi == 1) {
                    outF[idx] = val + res[idx];
                } else {
                    float u = val;
                    float c = 0.7978845608028654f * (u + 0.044715f * u * u * u);
                    float gl = 0.5f * u * (1.0f + tanhf(c));
                    outB[idx] = (bf16_t)gl;
                }
            }
        }
    }
}

// ---------------------------------------------------------------------------
// Fused attention: one wave per (b, head, 16-row tile). scores = q k^T / 8
// via WMMA into LDS, softmax over M=640 in LDS, then P @ v via WMMA.
// ---------------------------------------------------------------------------
__global__ __launch_bounds__(32)
void attn_kernel(const bf16_t* __restrict__ q, const bf16_t* __restrict__ kmat,
                 const bf16_t* __restrict__ vmat, bf16_t* __restrict__ o) {
    __shared__ float sc[16][MKV];        // 40 KB
    __shared__ float invb[16];

    int bid = blockIdx.x;                // b*(16*128) + h*128 + rt
    int b  = bid >> 11;
    int hh = (bid >> 7) & 15;
    int rt = bid & 127;
    int row0 = rt * 16;

    int lane = threadIdx.x;
    int row = lane & 15, half = lane >> 4;

    // q A-fragments for K = 0..31 and 32..63 of this head
    Frag16 aq0, aq1;
    {
        const bf16_t* qr = q + ((size_t)(b * SEQL + row0 + row)) * DMODEL + hh * HDIM;
        int kb = half * 8;
        aq0.u2[0] = *(const u32x2*)(qr + kb);
        aq0.u2[1] = *(const u32x2*)(qr + kb + 4);
        aq0.u2[2] = *(const u32x2*)(qr + kb + 16);
        aq0.u2[3] = *(const u32x2*)(qr + kb + 20);
        const bf16_t* qr1 = qr + 32;
        aq1.u2[0] = *(const u32x2*)(qr1 + kb);
        aq1.u2[1] = *(const u32x2*)(qr1 + kb + 4);
        aq1.u2[2] = *(const u32x2*)(qr1 + kb + 16);
        aq1.u2[3] = *(const u32x2*)(qr1 + kb + 20);
    }

    int kkB = half * 8;
    for (int mt = 0; mt < MKV / 16; ++mt) {       // 40 column tiles of scores
        Frag16 fb0, fb1;
        const bf16_t* kr = kmat + ((size_t)(b * MKV + mt * 16 + row)) * DMODEL + hh * HDIM + kkB;
        fb0.u4[0] = *(const u32x4*)kr;
        fb0.u4[1] = *(const u32x4*)(kr + 16);
        fb1.u4[0] = *(const u32x4*)(kr + 32);
        fb1.u4[1] = *(const u32x4*)(kr + 48);
        v8f acc = {};
        acc = wmma_bf16(aq0.v, fb0.v, acc);
        acc = wmma_bf16(aq1.v, fb1.v, acc);
#pragma unroll
        for (int r = 0; r < 8; ++r)
            sc[half * 8 + r][mt * 16 + row] = acc[r] * 0.125f;   // 1/sqrt(64)
    }
    __syncthreads();

    // softmax over 640, lane halves split the columns, combine via shfl_xor 16
    {
        float mx = -1e30f;
        int base = half * (MKV / 2);
        for (int j = 0; j < MKV / 2; ++j) mx = fmaxf(mx, sc[row][base + j]);
        mx = fmaxf(mx, __shfl_xor(mx, 16, 32));
        float sum = 0.0f;
        for (int j = 0; j < MKV / 2; ++j) {
            float e = __expf(sc[row][base + j] - mx);
            sc[row][base + j] = e;
            sum += e;
        }
        sum += __shfl_xor(sum, 16, 32);
        if (half == 0) invb[row] = 1.0f / sum;
    }
    __syncthreads();

    // O = P @ V : 20 K-chunks of 32, 4 N-tiles of 16 (d = 0..63)
    v8f oacc[4] = {};
    for (int kc = 0; kc < MKV / 32; ++kc) {
        Frag16 pa;
        int kbp = half * 8 + kc * 32;
#pragma unroll
        for (int e = 0; e < 16; ++e) {
            int vv = e >> 1, hb = e & 1;
            int kidx = kbp + ((vv & 3) * 2 + hb) + ((vv >= 4) ? 16 : 0);
            pa.e[e] = (bf16_t)sc[row][kidx];
        }
        int mbase = half * 8 + kc * 32;
#pragma unroll
        for (int nt = 0; nt < 4; ++nt) {
            Frag16 pb;
            int n = hh * HDIM + nt * 16 + row;
#pragma unroll
            for (int e = 0; e < 16; ++e)
                pb.e[e] = vmat[((size_t)(b * MKV + mbase + e + ((e & 8) ? 8 : 0))) * DMODEL + n];
            oacc[nt] = wmma_bf16(pa.v, pb.v, oacc[nt]);
        }
    }

#pragma unroll
    for (int nt = 0; nt < 4; ++nt) {
        int col = hh * HDIM + nt * 16 + row;
#pragma unroll
        for (int r = 0; r < 8; ++r) {
            int M = half * 8 + r;
            float val = oacc[nt][r] * invb[M];
            o[((size_t)(b * SEQL + row0 + M)) * DMODEL + col] = (bf16_t)val;
        }
    }
}

// ---------------------------------------------------------------------------
// Launcher
// ---------------------------------------------------------------------------
extern "C" void kernel_launch(void* const* d_in, const int* in_sizes, int n_in,
                              void* d_out, int out_size, void* d_ws, size_t ws_size,
                              hipStream_t stream) {
    (void)in_sizes; (void)n_in; (void)out_size; (void)ws_size;

    const float* x     = (const float*)d_in[0];
    const float* gu    = (const float*)d_in[1];
    const float* ln1g  = (const float*)d_in[2];
    const float* ln1b  = (const float*)d_in[3];
    const float* ln2g  = (const float*)d_in[4];
    const float* ln2b  = (const float*)d_in[5];
    const float* wsal  = (const float*)d_in[6];
    const float* bsal  = (const float*)d_in[7];
    const float* inw   = (const float*)d_in[8];
    const float* inb   = (const float*)d_in[9];
    const float* outw  = (const float*)d_in[10];
    const float* outb  = (const float*)d_in[11];
    const float* wfc   = (const float*)d_in[12];
    const float* bfc   = (const float*)d_in[13];
    const float* wpj   = (const float*)d_in[14];
    const float* bpj   = (const float*)d_in[15];

    const size_t TOK = (size_t)BATCH * SEQL;          // 8192
    char* wsb = (char*)d_ws;
    size_t off = 0;
    bf16_t* h_bf   = (bf16_t*)(wsb + off); off += TOK * DMODEL * 2;            // 16 MB
    float*  sal    = (float*) (wsb + off); off += TOK * 4;                     // 32 KB
    int*    startp = (int*)   (wsb + off); off += 256;
    bf16_t* kv_bf  = (bf16_t*)(wsb + off); off += (size_t)BATCH * MKV * DMODEL * 2;  // 5 MB
    bf16_t* q_bf   = (bf16_t*)(wsb + off); off += TOK * DMODEL * 2;            // 16 MB
    bf16_t* k_bf   = (bf16_t*)(wsb + off); off += (size_t)BATCH * MKV * DMODEL * 2;
    bf16_t* v_bf   = (bf16_t*)(wsb + off); off += (size_t)BATCH * MKV * DMODEL * 2;
    bf16_t* o_bf   = (bf16_t*)(wsb + off); off += TOK * DMODEL * 2;            // 16 MB
    float*  x2     = (float*) (wsb + off); off += TOK * DMODEL * 4;            // 32 MB
    bf16_t* h2_bf  = (bf16_t*)(wsb + off); off += TOK * DMODEL * 2;            // 16 MB
    bf16_t* m1_bf  = (bf16_t*)(wsb + off); off += TOK * DFFN * 2;              // 64 MB
    bf16_t* wqkv_bf= (bf16_t*)(wsb + off); off += (size_t)3 * DMODEL * DMODEL * 2;
    bf16_t* wout_bf= (bf16_t*)(wsb + off); off += (size_t)DMODEL * DMODEL * 2;
    bf16_t* wfc_t  = (bf16_t*)(wsb + off); off += (size_t)DMODEL * DFFN * 2;
    bf16_t* wpj_t  = (bf16_t*)(wsb + off); off += (size_t)DFFN * DMODEL * 2;

    // 1) weights -> bf16 (qkv/out keep [N][K] layout; fc/proj transposed to it)
    cvt_bf_kernel<<<2048, 256, 0, stream>>>(inw,  wqkv_bf, 3 * DMODEL * DMODEL);
    cvt_bf_kernel<<<1024, 256, 0, stream>>>(outw, wout_bf, DMODEL * DMODEL);
    cvt_tr_kernel<<<2048, 256, 0, stream>>>(wfc, wfc_t, DMODEL, DFFN);
    cvt_tr_kernel<<<2048, 256, 0, stream>>>(wpj, wpj_t, DFFN, DMODEL);

    // 2) LN1 (+ fused salience), 3) saccade selection, 4) gather windows
    ln_kernel<<<(int)TOK, 256, 0, stream>>>(x, ln1g, ln1b, wsal, h_bf, sal, 1);
    select_kernel<<<1, 256, 0, stream>>>(sal, gu, bsal, startp);
    gather_kernel<<<BATCH * MKV, 128, 0, stream>>>(h_bf, startp, kv_bf);

    // 5) Q/K/V projections (bf16 WMMA GEMMs, TDM tile staging)
    dim3 gq(DMODEL / 128, (int)TOK / 128);                 // 8 x 64
    gemm_kernel<<<gq, 256, 0, stream>>>(h_bf, DMODEL, wqkv_bf, inb,
                                        nullptr, nullptr, q_bf,
                                        (int)TOK, DMODEL, DMODEL, DMODEL, 0);
    dim3 gkv(DMODEL / 128, BATCH * MKV / 128);             // 8 x 20
    gemm_kernel<<<gkv, 256, 0, stream>>>(kv_bf, DMODEL, wqkv_bf + (size_t)DMODEL * DMODEL,
                                         inb + DMODEL, nullptr, nullptr, k_bf,
                                         BATCH * MKV, DMODEL, DMODEL, DMODEL, 0);
    gemm_kernel<<<gkv, 256, 0, stream>>>(kv_bf, DMODEL, wqkv_bf + (size_t)2 * DMODEL * DMODEL,
                                         inb + 2 * DMODEL, nullptr, nullptr, v_bf,
                                         BATCH * MKV, DMODEL, DMODEL, DMODEL, 0);

    // 6) fused attention (scores + softmax + P@V, all WMMA)
    attn_kernel<<<BATCH * NHEAD * (SEQL / 16), 32, 0, stream>>>(q_bf, k_bf, v_bf, o_bf);

    // 7) out projection + residual -> x2 (fp32)
    gemm_kernel<<<gq, 256, 0, stream>>>(o_bf, DMODEL, wout_bf, outb,
                                        x, x2, nullptr,
                                        (int)TOK, DMODEL, DMODEL, DMODEL, 1);

    // 8) LN2 -> h2 (bf16)
    ln_kernel<<<(int)TOK, 256, 0, stream>>>(x2, ln2g, ln2b, nullptr, h2_bf, nullptr, 0);

    // 9) MLP: FC + GELU -> m1 (bf16), then proj + bias + residual -> d_out (fp32)
    dim3 gfc(DFFN / 128, (int)TOK / 128);                  // 32 x 64
    gemm_kernel<<<gfc, 256, 0, stream>>>(h2_bf, DMODEL, wfc_t, bfc,
                                         nullptr, nullptr, m1_bf,
                                         (int)TOK, DFFN, DMODEL, DFFN, 2);
    gemm_kernel<<<gq, 256, 0, stream>>>(m1_bf, DFFN, wpj_t, bpj,
                                        x2, (float*)d_out, nullptr,
                                        (int)TOK, DMODEL, DFFN, DMODEL, 1);
}
